// MultiHeadAttention_11562051961554
// MI455X (gfx1250) — matrix operand order repaired and checked by reading, not compiled, not run on hardware
//
#include <hip/hip_runtime.h>

// ---------------------------------------------------------------------------
// MHA with RoPE for MI455X (gfx1250, wave32, WMMA bf16, TDM async tile loads).
// ---------------------------------------------------------------------------

typedef __attribute__((ext_vector_type(16))) __bf16        v16bf;
typedef __attribute__((ext_vector_type(8)))  float         v8f;
typedef __attribute__((ext_vector_type(4)))  unsigned int  u32x4;
typedef __attribute__((ext_vector_type(8)))  int           i32x8;
typedef __attribute__((ext_vector_type(4)))  int           i32x4;

#define BATCH    4
#define SEQ      2048
#define DMODEL   2048
#define NHEADS   16
#define HEADDIM  128
#define GEMM_M   (BATCH * SEQ)   // 8192

#define HAVE_TDM __has_builtin(__builtin_amdgcn_tensor_load_to_lds)

union BFrag { uint4 q[2]; v16bf v; };

__device__ __forceinline__ unsigned short f2bf(float f) {
    unsigned int u = __float_as_uint(f);
    u += 0x7fffu + ((u >> 16) & 1u);          // round-to-nearest-even
    return (unsigned short)(u >> 16);
}

// LDS byte offset of a __shared__ object (flat -> local, then ptrtoint).
__device__ __forceinline__ unsigned lds_addr_of(const void* p) {
    return (unsigned)(unsigned long long)(__attribute__((address_space(3))) const void*)p;
}

#if HAVE_TDM
// Issue one TDM 2D tile load: bf16 tile [128 rows x 32 cols] from a row-major
// tensor (row stride DMODEL elements) into LDS at lds_byte_addr.
__device__ __forceinline__ void tdm_load_tile_bf16(unsigned lds_byte_addr,
                                                   const unsigned short* gbase,
                                                   unsigned row0, unsigned k0,
                                                   unsigned tensor_rows)
{
    unsigned long long ga = (unsigned long long)gbase
                          + 2ull * ((unsigned long long)row0 * DMODEL + k0);
    u32x4 g0;
    g0.x = 1u;                                                 // count=1 (valid)
    g0.y = lds_byte_addr;                                      // lds_addr
    g0.z = (unsigned)(ga & 0xffffffffu);                       // global_addr[31:0]
    g0.w = (unsigned)((ga >> 32) & 0x01ffffffu) | (2u << 30);  // addr[56:32] | type=2
    i32x8 g1;
    g1[0] = 0x00010000;                                        // data_size=1 (2B), mask=0
    g1[1] = (int)((unsigned)DMODEL << 16);                     // tensor_dim0[15:0]<<16
    g1[2] = (int)(((unsigned)DMODEL >> 16) | ((tensor_rows & 0xffffu) << 16));
    g1[3] = (int)((tensor_rows >> 16) | (32u << 16));          // tile_dim0 = 32
    g1[4] = 128;                                               // tile_dim1=128, tile_dim2=0
    g1[5] = DMODEL;                                            // tensor_dim0_stride = 2048
    g1[6] = 0;
    g1[7] = 0;
    i32x4 z4 = (i32x4){0, 0, 0, 0};
    i32x8 z8 = (i32x8){0, 0, 0, 0, 0, 0, 0, 0};
    // amdgpu-toolchain (clang-23) 6-arg form: (g0, g1, g2, g3, g4, cpol)
    __builtin_amdgcn_tensor_load_to_lds(g0, g1, z4, z4, z8, 0);
}
#endif

// ---------------------------------------------------------------------------
// One-shot f32 -> bf16 conversion (8 elements / thread).
// ---------------------------------------------------------------------------
__global__ __launch_bounds__(256)
void f32_to_bf16(const float* __restrict__ src, unsigned short* __restrict__ dst)
{
    size_t i = (size_t)blockIdx.x * 256 + threadIdx.x;
    const float4* s = (const float4*)src + i * 2;
    float4 f0 = s[0], f1 = s[1];
    unsigned short t[8] = { f2bf(f0.x), f2bf(f0.y), f2bf(f0.z), f2bf(f0.w),
                            f2bf(f1.x), f2bf(f1.y), f2bf(f1.z), f2bf(f1.w) };
    ((uint4*)dst)[i] = *(const uint4*)t;
}

// ---------------------------------------------------------------------------
// GEMM: C[M,N] = A[M,K] @ W[N,K]^T, all-bf16 inputs, f32 accumulate.
// Block tile 128x128, 8 waves (2x4), K-step 32, TDM double-buffered LDS.
// MODE 0: Q -> bf16 [b,h,s,dh], RoPE + 1/sqrt(dh)
// MODE 1: K -> bf16 [b,h,s,dh], RoPE
// MODE 2: V -> bf16 [b,h,dh,s] (transposed for flash V B-operand)
// MODE 3: out f32 row-major [M,N] (O-projection)
// ---------------------------------------------------------------------------
template<int MODE>
__global__ __launch_bounds__(256)
void gemm_wmma(const unsigned short* __restrict__ A16,
               const unsigned short* __restrict__ W16,
               const int* __restrict__ tpos,
               unsigned short* __restrict__ out16,
               float* __restrict__ out32)
{
    __shared__ unsigned short As[2][128 * 32];
    __shared__ unsigned short Ws[2][128 * 32];

    const int tid   = threadIdx.x;
    const int lane  = tid & 31;
    const int wid   = tid >> 5;     // 0..7
    const int waveM = wid >> 2;     // 0..1 (64 rows each)
    const int waveN = wid & 3;      // 0..3 (32 cols each)
    const int bM    = blockIdx.y * 128;
    const int bN    = blockIdx.x * 128;
    const int half  = lane >> 4;
    const int lr    = lane & 15;

    v8f acc[4][2];
    for (int i = 0; i < 4; ++i)
        for (int j = 0; j < 2; ++j) acc[i][j] = (v8f){};

    const int NT = DMODEL / 32;     // 64 K-steps

#if HAVE_TDM
    unsigned asAddr[2] = { lds_addr_of(As[0]), lds_addr_of(As[1]) };
    unsigned wsAddr[2] = { lds_addr_of(Ws[0]), lds_addr_of(Ws[1]) };
    if (tid < 32) {   // wave 0 drives the Tensor Data Mover
        tdm_load_tile_bf16(asAddr[0], A16, bM, 0, GEMM_M);
        tdm_load_tile_bf16(wsAddr[0], W16, bN, 0, DMODEL);
    }
    for (int t = 0; t < NT; ++t) {
        const int buf = t & 1;
        if (tid < 32) __builtin_amdgcn_s_wait_tensorcnt((short)0);
        __syncthreads();                         // tile[buf] visible to all waves
        if (tid < 32 && (t + 1) < NT) {          // prefetch next tile via TDM
            int k1 = (t + 1) * 32;
            tdm_load_tile_bf16(asAddr[buf ^ 1], A16, bM, k1, GEMM_M);
            tdm_load_tile_bf16(wsAddr[buf ^ 1], W16, bN, k1, DMODEL);
        }
        BFrag a[4];
        for (int i = 0; i < 4; ++i) {
            const unsigned short* base = &As[buf][(waveM * 64 + i * 16 + lr) * 32];
            a[i].q[0] = *(const uint4*)(base + half * 8);
            a[i].q[1] = *(const uint4*)(base + half * 8 + 16);
        }
        BFrag b[2];
        for (int j = 0; j < 2; ++j) {
            const unsigned short* base = &Ws[buf][(waveN * 32 + j * 16 + lr) * 32];
            b[j].q[0] = *(const uint4*)(base + half * 8);
            b[j].q[1] = *(const uint4*)(base + half * 8 + 16);
        }
        for (int i = 0; i < 4; ++i)
            for (int j = 0; j < 2; ++j)
                acc[i][j] = __builtin_amdgcn_wmma_f32_16x16x32_bf16(
                    false, a[i].v, false, b[j].v, (short)0, acc[i][j], false, false);
        __syncthreads();                         // reads done before TDM reuses buffer
    }
#else
    // Fallback: manual staging (single buffer), 16 bf16 (2 rows x 8) per thread per operand tile.
    const int r0 = tid >> 2;
    const int cg = (tid & 3) * 8;
    for (int t = 0; t < NT; ++t) {
        int k0 = t * 32;
        for (int rr = 0; rr < 128; rr += 64) {
            int row = r0 + rr;
            *(uint4*)(&As[0][row * 32 + cg]) =
                *(const uint4*)(A16 + (size_t)(bM + row) * DMODEL + k0 + cg);
            *(uint4*)(&Ws[0][row * 32 + cg]) =
                *(const uint4*)(W16 + (size_t)(bN + row) * DMODEL + k0 + cg);
        }
        __syncthreads();
        BFrag a[4];
        for (int i = 0; i < 4; ++i) {
            const unsigned short* base = &As[0][(waveM * 64 + i * 16 + lr) * 32];
            a[i].q[0] = *(const uint4*)(base + half * 8);
            a[i].q[1] = *(const uint4*)(base + half * 8 + 16);
        }
        BFrag b[2];
        for (int j = 0; j < 2; ++j) {
            const unsigned short* base = &Ws[0][(waveN * 32 + j * 16 + lr) * 32];
            b[j].q[0] = *(const uint4*)(base + half * 8);
            b[j].q[1] = *(const uint4*)(base + half * 8 + 16);
        }
        for (int i = 0; i < 4; ++i)
            for (int j = 0; j < 2; ++j)
                acc[i][j] = __builtin_amdgcn_wmma_f32_16x16x32_bf16(
                    false, a[i].v, false, b[j].v, (short)0, acc[i][j], false, false);
        __syncthreads();
    }
#endif

    // ---- epilogue ----
    for (int i = 0; i < 4; ++i) {
        for (int j = 0; j < 2; ++j) {
            int mbase = bM + waveM * 64 + i * 16 + half * 8;
            int nn    = bN + waveN * 32 + j * 16 + lr;
            for (int r = 0; r < 8; ++r) {
                int   m = mbase + r;
                float v = acc[i][j][r];
                if (MODE <= 1) {
                    int s  = m & (SEQ - 1);
                    int dh = nn & (HEADDIM - 1);
                    int h  = nn >> 7;
                    int bi = m >> 11;
                    float fi  = (float)(dh >> 1);
                    float ang = (float)tpos[s] * __powf(10000.0f, -fi * (1.0f / 64.0f));
                    float sn, c; __sincosf(ang, &sn, &c);
                    float vp = __shfl_xor(v, 1);   // even/odd dh pairs = adjacent lanes
                    float res = ((lane & 1) == 0) ? (c * v - sn * vp)
                                                  : (sn * vp + c * v);
                    if (MODE == 0) res *= 0.08838834764831845f;  // 1/sqrt(128)
                    out16[((size_t)(bi * NHEADS + h) * SEQ + s) * HEADDIM + dh] = f2bf(res);
                } else if (MODE == 2) {
                    int s  = m & (SEQ - 1);
                    int dh = nn & (HEADDIM - 1);
                    int h  = nn >> 7;
                    int bi = m >> 11;
                    out16[((size_t)(bi * NHEADS + h) * HEADDIM + dh) * SEQ + s] = f2bf(v);
                } else {
                    out32[(size_t)m * DMODEL + nn] = v;
                }
            }
        }
    }
}

// ---------------------------------------------------------------------------
// Flash attention: each wave owns a 16-query tile for one (b,h).
// Causal key blocks of 32; QK^T and PV through WMMA bf16; online softmax.
// ---------------------------------------------------------------------------
__global__ __launch_bounds__(128)
void flash_attn(const unsigned short* __restrict__ Qb,
                const unsigned short* __restrict__ Kb,
                const unsigned short* __restrict__ Vt,
                unsigned short* __restrict__ Ob)
{
    __shared__ unsigned short Pl[4 * 16 * 32];   // per-wave P staging (1 KB each)
    const int lane = threadIdx.x & 31;
    const int wIn  = threadIdx.x >> 5;
    const int wg   = blockIdx.x * 4 + wIn;
    const int qt   = wg & (SEQ / 16 - 1);
    const int bh   = wg >> 7;
    const int q0   = qt * 16;

    const unsigned short* Qp = Qb + (size_t)bh * SEQ * HEADDIM;
    const unsigned short* Kp = Kb + (size_t)bh * SEQ * HEADDIM;
    const unsigned short* Vp = Vt + (size_t)bh * HEADDIM * SEQ;
    unsigned short* Pw = Pl + wIn * (16 * 32);

    const int half = lane >> 4;
    const int lr   = lane & 15;

    BFrag aq[4];
    for (int c = 0; c < 4; ++c) {
        const unsigned short* base = Qp + (size_t)(q0 + lr) * HEADDIM + c * 32;
        aq[c].q[0] = *(const uint4*)(base + half * 8);
        aq[c].q[1] = *(const uint4*)(base + half * 8 + 16);
    }

    v8f o[8];
    for (int c = 0; c < 8; ++c) o[c] = (v8f){};
    float mrow[8], lrow[8];
    for (int r = 0; r < 8; ++r) { mrow[r] = -1e30f; lrow[r] = 0.0f; }

    const int kbEnd = q0 + 16;
    for (int kb = 0; kb < kbEnd; kb += 32) {
        v8f sc[2];
        sc[0] = (v8f){}; sc[1] = (v8f){};
        for (int t = 0; t < 2; ++t) {
            const unsigned short* kbase = Kp + (size_t)(kb + t * 16 + lr) * HEADDIM;
            for (int c = 0; c < 4; ++c) {
                BFrag kf;
                kf.q[0] = *(const uint4*)(kbase + c * 32 + half * 8);
                kf.q[1] = *(const uint4*)(kbase + c * 32 + half * 8 + 16);
                sc[t] = __builtin_amdgcn_wmma_f32_16x16x32_bf16(
                    false, aq[c].v, false, kf.v, (short)0, sc[t], false, false);
            }
        }
        for (int t = 0; t < 2; ++t) {
            int kcol = kb + t * 16 + lr;
            for (int r = 0; r < 8; ++r) {
                int q = q0 + half * 8 + r;
                if (kcol > q) sc[t][r] = -1e30f;
            }
        }
        for (int r = 0; r < 8; ++r) {
            float mx = fmaxf(sc[0][r], sc[1][r]);
            for (int off = 1; off < 16; off <<= 1) mx = fmaxf(mx, __shfl_xor(mx, off));
            float mn  = fmaxf(mrow[r], mx);
            float fac = __expf(mrow[r] - mn);
            float p0  = __expf(sc[0][r] - mn);
            float p1  = __expf(sc[1][r] - mn);
            sc[0][r] = p0; sc[1][r] = p1;
            float rs = p0 + p1;
            for (int off = 1; off < 16; off <<= 1) rs += __shfl_xor(rs, off);
            lrow[r] = lrow[r] * fac + rs;
            mrow[r] = mn;
            for (int c = 0; c < 8; ++c) o[c][r] *= fac;
        }
        for (int t = 0; t < 2; ++t)
            for (int r = 0; r < 8; ++r)
                Pw[(half * 8 + r) * 32 + t * 16 + lr] = f2bf(sc[t][r]);
        asm volatile("s_wait_dscnt 0" ::: "memory");   // order LDS write -> read
        BFrag pa;
        {
            const unsigned short* base = Pw + lr * 32;
            pa.q[0] = *(const uint4*)(base + half * 8);
            pa.q[1] = *(const uint4*)(base + half * 8 + 16);
        }
        for (int c = 0; c < 8; ++c) {
            const unsigned short* vbase = Vp + (size_t)(c * 16 + lr) * SEQ + kb;
            BFrag vf;
            vf.q[0] = *(const uint4*)(vbase + half * 8);
            vf.q[1] = *(const uint4*)(vbase + half * 8 + 16);
            o[c] = __builtin_amdgcn_wmma_f32_16x16x32_bf16(
                false, pa.v, false, vf.v, (short)0, o[c], false, false);
        }
    }

    const int b = bh >> 4;
    const int h = bh & 15;
    for (int r = 0; r < 8; ++r) {
        float inv = 1.0f / lrow[r];
        int s = q0 + half * 8 + r;
        size_t base = ((size_t)b * SEQ + s) * DMODEL + h * HEADDIM;
        for (int c = 0; c < 8; ++c)
            Ob[base + c * 16 + lr] = f2bf(o[c][r] * inv);
    }
}

// ---------------------------------------------------------------------------
extern "C" void kernel_launch(void* const* d_in, const int* in_sizes, int n_in,
                              void* d_out, int out_size, void* d_ws, size_t ws_size,
                              hipStream_t stream)
{
    const float* x  = (const float*)d_in[0];
    const int*   tp = (const int*)  d_in[1];
    const float* Wq = (const float*)d_in[2];
    const float* Wk = (const float*)d_in[3];
    const float* Wv = (const float*)d_in[4];
    const float* Wo = (const float*)d_in[5];
    float* out = (float*)d_out;

    const size_t xe = (size_t)GEMM_M * DMODEL;         // 16M elems
    const size_t we = (size_t)DMODEL * DMODEL;         // 4M elems
    unsigned short* xb  = (unsigned short*)d_ws;       // x bf16 (reused as Ob later)
    unsigned short* wqb = xb  + xe;
    unsigned short* wkb = wqb + we;
    unsigned short* wvb = wkb + we;
    unsigned short* wob = wvb + we;
    unsigned short* Qb  = wob + we;
    unsigned short* Kb  = Qb  + xe;
    unsigned short* Vt  = Kb  + xe;
    unsigned short* Ob  = xb;                          // alias: x dead after QKV GEMMs

    // f32 -> bf16 (8 elems/thread)
    f32_to_bf16<<<dim3(xe / 2048), dim3(256), 0, stream>>>(x,  xb);
    f32_to_bf16<<<dim3(we / 2048), dim3(256), 0, stream>>>(Wq, wqb);
    f32_to_bf16<<<dim3(we / 2048), dim3(256), 0, stream>>>(Wk, wkb);
    f32_to_bf16<<<dim3(we / 2048), dim3(256), 0, stream>>>(Wv, wvb);
    f32_to_bf16<<<dim3(we / 2048), dim3(256), 0, stream>>>(Wo, wob);

    dim3 ggrid(DMODEL / 128, GEMM_M / 128);
    dim3 gblock(256);
    gemm_wmma<0><<<ggrid, gblock, 0, stream>>>(xb, wqb, tp, Qb, nullptr);
    gemm_wmma<1><<<ggrid, gblock, 0, stream>>>(xb, wkb, tp, Kb, nullptr);
    gemm_wmma<2><<<ggrid, gblock, 0, stream>>>(xb, wvb, tp, Vt, nullptr);

    flash_attn<<<dim3((BATCH * NHEADS * SEQ / 16) / 4), dim3(128), 0, stream>>>(Qb, Kb, Vt, Ob);

    gemm_wmma<3><<<ggrid, gblock, 0, stream>>>(Ob, wob, tp, nullptr, out);
}
